// Net_11192684773569
// MI455X (gfx1250) — matrix operand order repaired and checked by reading, not compiled, not run on hardware
//
#include <hip/hip_runtime.h>
#include <hip/hip_bf16.h>
#include <math.h>

// ---------------------------------------------------------------------------
// Model dimensions (fixed by the reference)
// ---------------------------------------------------------------------------
#define BB   8
#define LL   512
#define MTOK (BB * LL)        // 4096 tokens
#define INF_ 64
#define HH   512
#define ED_  1024
#define NN   16
#define DC_  4
#define DTR_ 32
#define LYR_ 4

typedef __attribute__((ext_vector_type(16))) __bf16 v16bf;
typedef __attribute__((ext_vector_type(8)))  __bf16 v8bf;
typedef __attribute__((ext_vector_type(8)))  float  v8f;

// ---------------------------------------------------------------------------
// Device helpers
// ---------------------------------------------------------------------------
__device__ __forceinline__ unsigned short f2bf(float f) {
    union { float f; unsigned int u; } v; v.f = f;
    unsigned int u = v.u;
    u += 0x7fffu + ((u >> 16) & 1u);          // round-to-nearest-even
    return (unsigned short)(u >> 16);
}
__device__ __forceinline__ float geluf(float x) {
    return 0.5f * x * (1.f + erff(x * 0.70710678118654752f));
}
__device__ __forceinline__ float siluf(float x) {
    return x / (1.f + __expf(-x));
}
__device__ __forceinline__ float softplusf(float x) {
    return (x > 20.f) ? x : log1pf(__expf(x));
}

// ---------------------------------------------------------------------------
// f32 -> bf16 conversion
// ---------------------------------------------------------------------------
__global__ __launch_bounds__(256) void cvt_bf16_k(
    const float* __restrict__ src, unsigned short* __restrict__ dst, int n) {
    int i = blockIdx.x * 256 + threadIdx.x;
    if (i < n) dst[i] = f2bf(src[i]);
}

// ---------------------------------------------------------------------------
// WMMA GEMM:  C[M,N] = epi( A[M,lda(bf16)] * W[N,K(bf16)]^T + bias )
// One wave -> 32 x (16*NT) tile: 2 M-subtiles x NT N-subtiles of 16x16 WMMA.
// NT=4 -> 8 WMMAs per 12 b128 loads (A frags reused x4, B frags x2).
// EPI: 0=store  1=+bias  2=softplus(+bias)  3=gelu(+bias)  4=C += acc
// ---------------------------------------------------------------------------
template<int EPI, int NT>
__global__ __launch_bounds__(128) void gemm_wmma_bf16(
    const unsigned short* __restrict__ A,   // bf16 raw, row stride lda
    const unsigned short* __restrict__ W,   // bf16 raw, [N x K] row-major
    const float* __restrict__ bias,         // [N] (may be null for EPI 0/4)
    float* __restrict__ C,                  // f32, row stride ldc
    int M, int N, int K, int lda, int ldc)
{
    const int lane = threadIdx.x & 31;
    const int r    = lane & 15;
    const int hi   = lane >> 4;
    const int n0   = (blockIdx.x * 4 + threadIdx.y) * (16 * NT);
    const int m0   = blockIdx.y * 32;
    if (n0 >= N || m0 >= M) return;         // wave-uniform: EXEC stays all-1s

    union Frag { v16bf v; v8bf h[2]; };
    v8f acc[2][NT];
#pragma unroll
    for (int i = 0; i < 2; ++i)
#pragma unroll
        for (int j = 0; j < NT; ++j)
            acc[i][j] = (v8f){0.f,0.f,0.f,0.f,0.f,0.f,0.f,0.f};

    const unsigned short* pa[2];
    const unsigned short* pb[NT];
    pa[0] = A + (size_t)(m0 + r)      * lda;
    pa[1] = A + (size_t)(m0 + 16 + r) * lda;
#pragma unroll
    for (int j = 0; j < NT; ++j)
        pb[j] = W + (size_t)(n0 + j * 16 + r) * K;

    for (int k0 = 0; k0 < K; k0 += 32) {
        Frag a[2], b[NT];
        // A 16x32 bf16 layout: lanes hi=0 hold K {0..7,16..23}, hi=1 {8..15,24..31}
#pragma unroll
        for (int i = 0; i < 2; ++i) {
            a[i].h[0] = *(const v8bf*)(pa[i] + k0 + hi * 8);
            a[i].h[1] = *(const v8bf*)(pa[i] + k0 + 16 + hi * 8);
        }
        // B 32x16 bf16 layout: lanes hi=0 hold K {0..7,16..23}, hi=1 {8..15,24..31}
#pragma unroll
        for (int j = 0; j < NT; ++j) {
            b[j].h[0] = *(const v8bf*)(pb[j] + k0 + hi * 8);
            b[j].h[1] = *(const v8bf*)(pb[j] + k0 + 16 + hi * 8);
        }
        if (k0 + 32 < K) {                  // global_prefetch_b8 for next K-slab
            __builtin_prefetch(pa[0] + k0 + 32, 0, 1);
            __builtin_prefetch(pa[1] + k0 + 32, 0, 1);
            __builtin_prefetch(pb[0] + k0 + 32, 0, 1);
            __builtin_prefetch(pb[NT - 1] + k0 + 32, 0, 1);
        }
#pragma unroll
        for (int i = 0; i < 2; ++i)
#pragma unroll
            for (int j = 0; j < NT; ++j)
                acc[i][j] = __builtin_amdgcn_wmma_f32_16x16x32_bf16(
                    false, a[i].v, false, b[j].v, (short)0, acc[i][j], false, false);
    }

    float bv[NT];
#pragma unroll
    for (int j = 0; j < NT; ++j)
        bv[j] = (EPI == 1 || EPI == 2 || EPI == 3) ? bias[n0 + j * 16 + r] : 0.f;

    // C/D layout: lane r -> col, vgpr i + hi*8 -> row offset within 16x16 tile
#pragma unroll
    for (int ti = 0; ti < 2; ++ti) {
#pragma unroll
        for (int tj = 0; tj < NT; ++tj) {
            const int col = n0 + tj * 16 + r;
#pragma unroll
            for (int i = 0; i < 8; ++i) {
                int row = m0 + ti * 16 + hi * 8 + i;
                float v = acc[ti][tj][i];
                if (EPI == 1 || EPI == 2 || EPI == 3) v += bv[tj];
                if (EPI == 2) v = softplusf(v);
                if (EPI == 3) v = geluf(v);
                size_t idx = (size_t)row * ldc + col;
                if (EPI == 4) C[idx] += v; else C[idx] = v;
            }
        }
    }
}

// ---------------------------------------------------------------------------
// LayerNorm + exact GELU, one block per token row (H=512)
// ---------------------------------------------------------------------------
__global__ __launch_bounds__(256) void ln_gelu_k(
    const float* __restrict__ X, const float* __restrict__ w,
    const float* __restrict__ b, float* __restrict__ Y, int H)
{
    __shared__ float red[256];
    const int row = blockIdx.x, tid = threadIdx.x;
    const float* x = X + (size_t)row * H;

    float s = 0.f;
    for (int i = tid; i < H; i += 256) s += x[i];
    red[tid] = s; __syncthreads();
    for (int o = 128; o > 0; o >>= 1) { if (tid < o) red[tid] += red[tid + o]; __syncthreads(); }
    const float mu = red[0] / H;
    __syncthreads();

    float s2 = 0.f;
    for (int i = tid; i < H; i += 256) { float d = x[i] - mu; s2 += d * d; }
    red[tid] = s2; __syncthreads();
    for (int o = 128; o > 0; o >>= 1) { if (tid < o) red[tid] += red[tid + o]; __syncthreads(); }
    const float rstd = rsqrtf(red[0] / H + 1e-5f);

    float* y = Y + (size_t)row * H;
    for (int i = tid; i < H; i += 256)
        y[i] = geluf((x[i] - mu) * rstd * w[i] + b[i]);
}

// ---------------------------------------------------------------------------
// RMSNorm fused with f32->bf16 conversion, one block per token row
// ---------------------------------------------------------------------------
__global__ __launch_bounds__(256) void rms_bf16_k(
    const float* __restrict__ X, const float* __restrict__ w,
    unsigned short* __restrict__ Y, int H)
{
    __shared__ float red[256];
    const int row = blockIdx.x, tid = threadIdx.x;
    const float* x = X + (size_t)row * H;

    float s2 = 0.f;
    for (int i = tid; i < H; i += 256) { float v = x[i]; s2 += v * v; }
    red[tid] = s2; __syncthreads();
    for (int o = 128; o > 0; o >>= 1) { if (tid < o) red[tid] += red[tid + o]; __syncthreads(); }
    const float rinv = rsqrtf(red[0] / H + 1e-5f);

    unsigned short* y = Y + (size_t)row * H;
    for (int i = tid; i < H; i += 256) y[i] = f2bf(x[i] * rinv * w[i]);
}

// ---------------------------------------------------------------------------
// Causal depthwise conv (DC=4) + bias + SiLU.  xc = xz[:, :ED]
// ---------------------------------------------------------------------------
__global__ __launch_bounds__(256) void conv_silu_k(
    const float* __restrict__ xz, const float* __restrict__ cw,
    const float* __restrict__ cb, float* __restrict__ u, int total)
{
    int i = blockIdx.x * 256 + threadIdx.x;
    if (i >= total) return;
    const int e   = i & (ED_ - 1);
    const int tok = i >> 10;                 // ED_ == 1024
    const int l   = tok & (LL - 1);
    float s = cb[e];
    const float* cwe = cw + e * DC_;
#pragma unroll
    for (int j = 0; j < DC_; ++j) {
        int ll = l - (DC_ - 1) + j;
        if (ll >= 0) s += cwe[j] * xz[(size_t)(tok - (DC_ - 1) + j) * (2 * ED_) + e];
    }
    u[i] = siluf(s);
}

// ---------------------------------------------------------------------------
// Fused selective scan.  One thread per (batch, channel); 16-wide state in
// registers; per-timestep B/C vectors staged in LDS; epilogue fuses +D*u and
// *silu(z).  Writes y in place over u (read-before-write per thread).
// ---------------------------------------------------------------------------
__global__ __launch_bounds__(256) void ssm_scan_k(
    const float* __restrict__ delta, float* __restrict__ u,
    const float* __restrict__ dbc,   const float* __restrict__ xz,
    const float* __restrict__ A_log, const float* __restrict__ Dp)
{
    __shared__ float sBC[2 * NN];            // B[0..15], C[16..31]
    const int tid = threadIdx.x;
    const int b   = blockIdx.x >> 2;         // ED/256 = 4 blocks per batch
    const int e   = ((blockIdx.x & 3) << 8) + tid;

    float a[NN], h[NN];
#pragma unroll
    for (int n = 0; n < NN; ++n) { a[n] = -__expf(A_log[e * NN + n]); h[n] = 0.f; }
    const float Dv = Dp[e];
    const size_t tokBase = (size_t)b * LL;

    for (int l = 0; l < LL; ++l) {
        const size_t tok = tokBase + l;
        if (tid < 2 * NN) sBC[tid] = dbc[tok * (DTR_ + 2 * NN) + DTR_ + tid];
        __syncthreads();

        const float dt = delta[tok * ED_ + e];
        const float uu = u[tok * ED_ + e];
        const float zz = xz[tok * (2 * ED_) + ED_ + e];
        float acc = 0.f;
#pragma unroll
        for (int n = 0; n < NN; ++n) {
            h[n] = __expf(dt * a[n]) * h[n] + dt * sBC[n] * uu;
            acc  = fmaf(h[n], sBC[NN + n], acc);
        }
        __syncthreads();                     // protect sBC before next overwrite
        u[tok * ED_ + e] = (acc + Dv * uu) * siluf(zz);
    }
}

// ---------------------------------------------------------------------------
// Elementwise residual add: a += b
// ---------------------------------------------------------------------------
__global__ __launch_bounds__(256) void add_inplace_k(
    float* __restrict__ a, const float* __restrict__ b, int n) {
    int i = blockIdx.x * 256 + threadIdx.x;
    if (i < n) a[i] += b[i];
}

// ---------------------------------------------------------------------------
// Head: per-token dot(256) + bias + sigmoid, one wave32 per token
// ---------------------------------------------------------------------------
__global__ __launch_bounds__(256) void head_k(
    const float* __restrict__ X, const float* __restrict__ w,
    const float* __restrict__ b, float* __restrict__ out, int M)
{
    const int wave = threadIdx.x >> 5, lane = threadIdx.x & 31;
    const int row  = blockIdx.x * 8 + wave;
    if (row >= M) return;
    const float* x = X + (size_t)row * (HH / 2);
    float s = 0.f;
#pragma unroll
    for (int i = 0; i < 8; ++i) s = fmaf(x[lane + i * 32], w[lane + i * 32], s);
#pragma unroll
    for (int m = 16; m > 0; m >>= 1) s += __shfl_xor(s, m, 32);
    if (lane == 0) out[row] = 1.f / (1.f + __expf(-(s + b[0])));
}

// ---------------------------------------------------------------------------
// Host orchestration
// ---------------------------------------------------------------------------
extern "C" void kernel_launch(void* const* d_in, const int* in_sizes, int n_in,
                              void* d_out, int out_size, void* d_ws, size_t ws_size,
                              hipStream_t stream) {
    (void)in_sizes; (void)n_in; (void)out_size; (void)ws_size;

    const float* x        = (const float*)d_in[0];
    const float* w_in     = (const float*)d_in[1];
    const float* b_in     = (const float*)d_in[2];
    const float* ln1_w    = (const float*)d_in[3];
    const float* ln1_b    = (const float*)d_in[4];
    const float* norm_w   = (const float*)d_in[5];
    const float* in_projw = (const float*)d_in[6];
    const float* conv_w   = (const float*)d_in[7];
    const float* conv_b   = (const float*)d_in[8];
    const float* x_projw  = (const float*)d_in[9];
    const float* dt_projw = (const float*)d_in[10];
    const float* dt_projb = (const float*)d_in[11];
    const float* A_log    = (const float*)d_in[12];
    const float* Dp       = (const float*)d_in[13];
    const float* out_projw= (const float*)d_in[14];
    const float* w_ref    = (const float*)d_in[15];
    const float* b_ref    = (const float*)d_in[16];
    const float* ln2_w    = (const float*)d_in[17];
    const float* ln2_b    = (const float*)d_in[18];
    const float* w_o1     = (const float*)d_in[19];
    const float* b_o1     = (const float*)d_in[20];
    const float* w_o2     = (const float*)d_in[21];
    const float* b_o2     = (const float*)d_in[22];
    float* out = (float*)d_out;

    // ---- workspace layout (bytes, 256-B aligned) ----
    char* ws = (char*)d_ws;
    const size_t SZ_H   = (size_t)MTOK * HH * 4;       // 8 MB
    const size_t SZ_XZ  = (size_t)MTOK * 2 * ED_ * 4;  // 32 MB
    const size_t SZ_U   = (size_t)MTOK * ED_ * 4;      // 16 MB
    const size_t SZ_DBC = (size_t)MTOK * 64 * 4;       // 1 MB
    float*          f_h    = (float*)(ws);
    float*          f_hm   = (float*)(ws + SZ_H);
    float*          f_xz   = (float*)(ws + 2 * SZ_H);
    float*          f_u    = (float*)(ws + 2 * SZ_H + SZ_XZ);
    float*          f_dl   = (float*)(ws + 2 * SZ_H + SZ_XZ + SZ_U);
    float*          f_dbc  = (float*)(ws + 2 * SZ_H + SZ_XZ + 2 * SZ_U);
    float*          f_tmp  = (float*)(ws + 2 * SZ_H + SZ_XZ + 2 * SZ_U + SZ_DBC);
    unsigned short* bf_a   = (unsigned short*)(ws + 3 * SZ_H + SZ_XZ + 2 * SZ_U + SZ_DBC);
    unsigned short* bf_w   = (unsigned short*)(ws + 3 * SZ_H + SZ_XZ + 2 * SZ_U + SZ_DBC + SZ_U / 2);

    auto cvt = [&](const float* s, unsigned short* d, int n) {
        cvt_bf16_k<<<(n + 255) / 256, 256, 0, stream>>>(s, d, n);
    };
    auto gemm = [&](int epi, float* Cp, const unsigned short* Ap,
                    const unsigned short* Wp, const float* bp,
                    int M, int N, int K, int lda, int ldc) {
        if (N >= 256) {                     // 32x64 tile per wave, 8 WMMA / 12 loads
            dim3 grid((N + 255) / 256, M / 32), blk(32, 4);
            switch (epi) {
            case 0: gemm_wmma_bf16<0,4><<<grid, blk, 0, stream>>>(Ap, Wp, bp, Cp, M, N, K, lda, ldc); break;
            case 1: gemm_wmma_bf16<1,4><<<grid, blk, 0, stream>>>(Ap, Wp, bp, Cp, M, N, K, lda, ldc); break;
            case 2: gemm_wmma_bf16<2,4><<<grid, blk, 0, stream>>>(Ap, Wp, bp, Cp, M, N, K, lda, ldc); break;
            case 3: gemm_wmma_bf16<3,4><<<grid, blk, 0, stream>>>(Ap, Wp, bp, Cp, M, N, K, lda, ldc); break;
            default: gemm_wmma_bf16<4,4><<<grid, blk, 0, stream>>>(Ap, Wp, bp, Cp, M, N, K, lda, ldc); break;
            }
        } else {                            // narrow-N: 32x32 tile per wave
            dim3 grid((N + 127) / 128, M / 32), blk(32, 4);
            switch (epi) {
            case 0: gemm_wmma_bf16<0,2><<<grid, blk, 0, stream>>>(Ap, Wp, bp, Cp, M, N, K, lda, ldc); break;
            case 1: gemm_wmma_bf16<1,2><<<grid, blk, 0, stream>>>(Ap, Wp, bp, Cp, M, N, K, lda, ldc); break;
            case 2: gemm_wmma_bf16<2,2><<<grid, blk, 0, stream>>>(Ap, Wp, bp, Cp, M, N, K, lda, ldc); break;
            case 3: gemm_wmma_bf16<3,2><<<grid, blk, 0, stream>>>(Ap, Wp, bp, Cp, M, N, K, lda, ldc); break;
            default: gemm_wmma_bf16<4,2><<<grid, blk, 0, stream>>>(Ap, Wp, bp, Cp, M, N, K, lda, ldc); break;
            }
        }
    };

    // ---- embed: h = gelu(ln1(x @ w_in^T + b_in)) ----
    cvt(x, bf_a, MTOK * INF_);
    cvt(w_in, bf_w, HH * INF_);
    gemm(1, f_tmp, bf_a, bf_w, b_in, MTOK, HH, INF_, INF_, HH);
    ln_gelu_k<<<MTOK, 256, 0, stream>>>(f_tmp, ln1_w, ln1_b, f_h, HH);
    hipMemcpyAsync(f_hm, f_h, SZ_H, hipMemcpyDeviceToDevice, stream);

    // ---- 4 mixer layers, hm += mixer(rms(hm)) ----
    for (int l = 0; l < LYR_; ++l) {
        rms_bf16_k<<<MTOK, 256, 0, stream>>>(f_hm, norm_w + (size_t)l * HH, bf_a, HH);
        cvt(in_projw + (size_t)l * 2 * ED_ * HH, bf_w, 2 * ED_ * HH);
        gemm(0, f_xz, bf_a, bf_w, nullptr, MTOK, 2 * ED_, HH, HH, 2 * ED_);

        conv_silu_k<<<(MTOK * ED_) / 256, 256, 0, stream>>>(
            f_xz, conv_w + (size_t)l * ED_ * DC_, conv_b + (size_t)l * ED_, f_u, MTOK * ED_);

        cvt(f_u, bf_a, MTOK * ED_);
        cvt(x_projw + (size_t)l * 64 * ED_, bf_w, 64 * ED_);
        gemm(0, f_dbc, bf_a, bf_w, nullptr, MTOK, 64, ED_, ED_, 64);

        cvt(f_dbc, bf_a, MTOK * 64);
        cvt(dt_projw + (size_t)l * ED_ * DTR_, bf_w, ED_ * DTR_);
        gemm(2, f_dl, bf_a, bf_w, dt_projb + (size_t)l * ED_, MTOK, ED_, DTR_, 64, ED_);

        ssm_scan_k<<<BB * 4, 256, 0, stream>>>(
            f_dl, f_u, f_dbc, f_xz, A_log + (size_t)l * ED_ * NN, Dp + (size_t)l * ED_);

        cvt(f_u, bf_a, MTOK * ED_);
        cvt(out_projw + (size_t)l * HH * ED_, bf_w, HH * ED_);
        gemm(4, f_hm, bf_a, bf_w, nullptr, MTOK, HH, ED_, ED_, HH);   // hm += y @ W^T
    }

    // ---- h = h + hm; refine + head ----
    add_inplace_k<<<(MTOK * HH) / 256, 256, 0, stream>>>(f_h, f_hm, MTOK * HH);
    cvt(f_h, bf_a, MTOK * HH);
    cvt(w_ref, bf_w, HH * HH);
    gemm(1, f_tmp, bf_a, bf_w, b_ref, MTOK, HH, HH, HH, HH);
    ln_gelu_k<<<MTOK, 256, 0, stream>>>(f_tmp, ln2_w, ln2_b, f_hm, HH);
    cvt(f_hm, bf_a, MTOK * HH);
    cvt(w_o1, bf_w, (HH / 2) * HH);
    gemm(3, f_tmp, bf_a, bf_w, b_o1, MTOK, HH / 2, HH, HH, HH / 2);
    head_k<<<MTOK / 8, 256, 0, stream>>>(f_tmp, w_o2, b_o2, out, MTOK);
}
